// GPT_32882269618694
// MI455X (gfx1250) — matrix operand order repaired and checked by reading, not compiled, not run on hardware
//
#include <hip/hip_runtime.h>
#include <math.h>
#include <stdint.h>

// GPT forward for MI455X (gfx1250, wave32, WMMA, TDM).
// All GEMMs + attention run on v_wmma_f32_16x16x32_bf16 (f32 accumulate).
// Weights/GEMM-A activations are pre-cast to bf16 once per launch so the GEMM
// hot loop is pure data movement feeding WMMA. GEMM staging uses a
// double-buffered Tensor-Data-Mover pipeline (tensor_load_to_lds +
// s_wait_tensorcnt) so DMA of tile i+1 overlaps the WMMAs of tile i.
// Workspace requirement: ~257 MB.
// Input flattening assumption (dict/list insertion order):
//   0:input_ids 1:attention_mask(unused, causal recomputed) 2:emb_table 3:pos_enc
//   4+18*l+j : wq,bq,wk,bk,wv,bv,wo,bo,attn_ln_g,attn_ln_b,fc1,fc2,ff_ln_g,
//              ff_ln_b,rc1_g,rc1_b,rc2_g,rc2_b
//   112:proj_w 113:proj_b

#ifdef __has_builtin
#if __has_builtin(__builtin_amdgcn_tensor_load_to_lds)
#define HAVE_TDM 1
#endif
#endif
#ifndef HAVE_TDM
#define HAVE_TDM 0
#endif

namespace {

constexpr int D      = 512;
constexpr int S      = 2048;
constexpr int B      = 2;
constexpr int H      = 8;
constexpr int DH     = 64;
constexpr int FF     = 2048;
constexpr int LAYERS = 6;
constexpr int V      = 50257;
constexpr int M      = B * S;   // 4096 token rows

typedef __attribute__((ext_vector_type(16))) __bf16 bf16x16;
typedef __attribute__((ext_vector_type(2)))  __bf16 bf16x2;
typedef __attribute__((ext_vector_type(8)))  float  f32x8;
typedef __attribute__((ext_vector_type(4)))  unsigned int u32x4;
typedef __attribute__((ext_vector_type(8)))  int i32x8;
typedef __attribute__((ext_vector_type(4)))  int i32x4;

__device__ __forceinline__ f32x8 zero8() {
    f32x8 z = {0.f, 0.f, 0.f, 0.f, 0.f, 0.f, 0.f, 0.f};
    return z;
}

// Load a 16-element bf16 fragment as two 16-byte chunks (ds/global _load_b128).
// Offsets are in 2-byte elements.
__device__ __forceinline__ bf16x16 ldfrag(const void* row, int o0, int o1) {
    const unsigned short* p = (const unsigned short*)row;
    union { int4 q[2]; bf16x16 v; } u;
    u.q[0] = *(const int4*)(p + o0);
    u.q[1] = *(const int4*)(p + o1);
    return u.v;
}

__device__ __forceinline__ f32x8 wmma_bf16(bf16x16 a, bf16x16 b, f32x8 c) {
    // (neg_a, A, neg_b, B, c_mod, C, reuse_a, reuse_b)
    return __builtin_amdgcn_wmma_f32_16x16x32_bf16(false, a, false, b, (short)0, c,
                                                   false, false);
}

#if HAVE_TDM
__device__ __forceinline__ void wait_tensor0() {
#if __has_builtin(__builtin_amdgcn_s_wait_tensorcnt)
    __builtin_amdgcn_s_wait_tensorcnt(0);
#else
    asm volatile("s_wait_tensorcnt 0x0" ::: "memory");
#endif
}

__device__ __forceinline__ void wait_tensor2() {
#if __has_builtin(__builtin_amdgcn_s_wait_tensorcnt)
    __builtin_amdgcn_s_wait_tensorcnt(2);
#else
    asm volatile("s_wait_tensorcnt 0x2" ::: "memory");
#endif
}

// 2D TDM tile load, bf16 elements (data_size=1 -> 2 bytes).
// dim0/dim1 = remaining tensor extent from the tile origin (gives OOB zero-fill),
// stride0 = elements between rows, tile0 x tile1 = tile shape.
__device__ __forceinline__ void tdm_load_2d(unsigned lds_off, const void* gaddr,
                                            unsigned dim0, unsigned dim1,
                                            unsigned stride0,
                                            unsigned tile0, unsigned tile1) {
    unsigned long long ga = (unsigned long long)(uintptr_t)gaddr;
    u32x4 g0;
    g0[0] = 1u;                                         // count=1, user descriptor
    g0[1] = lds_off;                                    // LDS byte address
    g0[2] = (unsigned)(ga & 0xFFFFFFFFu);               // global_addr[31:0]
    g0[3] = (unsigned)((ga >> 32) & 0x1FFFFFFu)         // global_addr[56:32]
          | (2u << 30);                                 // type = 2 ("image")
    i32x8 g1;
    g1[0] = (int)(1u << 16);                            // data_size=1 (2B), no flags
    g1[1] = (int)((dim0 & 0xFFFFu) << 16);              // tensor_dim0[15:0]
    g1[2] = (int)(((dim0 >> 16) & 0xFFFFu) |            // tensor_dim0[31:16]
                  ((dim1 & 0xFFFFu) << 16));            // tensor_dim1[15:0]
    g1[3] = (int)(((dim1 >> 16) & 0xFFFFu) |            // tensor_dim1[31:16]
                  (tile0 << 16));                       // tile_dim0
    g1[4] = (int)(tile1 & 0xFFFFu);                     // tile_dim1 (tile_dim2=0)
    g1[5] = (int)stride0;                               // tensor_dim0_stride[31:0]
    g1[6] = 0;                                          // stride hi / dim1_stride lo
    g1[7] = 0;
    i32x4 gz4 = {0, 0, 0, 0};                           // groups 2/3 unused (2D)
    i32x8 gz8 = {0, 0, 0, 0, 0, 0, 0, 0};               // extra group (6-arg form)
    __builtin_amdgcn_tensor_load_to_lds(g0, g1, gz4, gz4, gz8, 0);
}
#endif

// ---------------------------------------------------------------------------
// GEMM: C[M x N] = A[M x K] @ W[N x K]^T (+ bias). A, W are bf16; acc f32.
// M multiple of 64, K multiple of 32. Block tile 64x128, 8 waves, each wave
// a 16x64 strip (4 wmma accumulators). TDM path double-buffers the LDS tiles
// so the DMA of tile i+1 overlaps the WMMAs of tile i.
// ---------------------------------------------------------------------------
template <typename TOut>
__global__ void __launch_bounds__(256)
gemm_kernel(const __bf16* __restrict__ A, const __bf16* __restrict__ W,
            const float* __restrict__ bias, TOut* __restrict__ C,
            int N, int K) {
    __shared__ unsigned short Al[2][64][32];     // 2 x 4 KB bf16 bits
    __shared__ unsigned short Bl[2][128][32];    // 2 x 8 KB bf16 bits

    const int tid  = threadIdx.x;
    const int wav  = tid >> 5;
    const int lane = tid & 31;
    const int hlf  = lane >> 4;
    const int l16  = lane & 15;
    const int wm   = wav & 3;     // M sub-tile: wm*16
    const int wn   = wav >> 2;    // N sub-tile: wn*64
    const int m0   = blockIdx.y * 64;
    const int n0   = blockIdx.x * 128;

    f32x8 acc[4];
    #pragma unroll
    for (int j = 0; j < 4; ++j) acc[j] = zero8();

#if HAVE_TDM
    const unsigned ldsA0 = (unsigned)(uintptr_t)&Al[0][0][0];
    const unsigned ldsA1 = (unsigned)(uintptr_t)&Al[1][0][0];
    const unsigned ldsB0 = (unsigned)(uintptr_t)&Bl[0][0][0];
    const unsigned ldsB1 = (unsigned)(uintptr_t)&Bl[1][0][0];
    const int nk = K >> 5;

    if (tid < 32) {   // wave 0 drives the Tensor Data Mover: prologue tile 0
        tdm_load_2d(ldsA0, A + (size_t)m0 * K,
                    (unsigned)K, (unsigned)(M - m0), (unsigned)K, 32u, 64u);
        tdm_load_2d(ldsB0, W + (size_t)n0 * K,
                    (unsigned)K, (unsigned)(N - n0), (unsigned)K, 32u, 128u);
    }
    for (int i = 0; i < nk; ++i) {
        const int cur = i & 1;
        if (tid < 32) {
            if (i + 1 < nk) {
                const int k1 = (i + 1) << 5;
                tdm_load_2d(cur ? ldsA0 : ldsA1, A + (size_t)m0 * K + k1,
                            (unsigned)(K - k1), (unsigned)(M - m0), (unsigned)K,
                            32u, 64u);
                tdm_load_2d(cur ? ldsB0 : ldsB1, W + (size_t)n0 * K + k1,
                            (unsigned)(K - k1), (unsigned)(N - n0), (unsigned)K,
                            32u, 128u);
                wait_tensor2();   // in-order TENSORcnt: tile i landed, i+1 in flight
            } else {
                wait_tensor0();
            }
        }
        __syncthreads();          // tile i visible to all waves

        // A fragment: M=l16, K chunks at hlf*8 and 16+hlf*8
        bf16x16 af = ldfrag(&Al[cur][wm * 16 + l16][0], hlf * 8, 16 + hlf * 8);
        #pragma unroll
        for (int j = 0; j < 4; ++j) {
            // B fragment: N=l16, K chunks at hlf*8 and 16+hlf*8
            bf16x16 bfv = ldfrag(&Bl[cur][wn * 64 + j * 16 + l16][0],
                                 hlf * 8, 16 + hlf * 8);
            acc[j] = wmma_bf16(af, bfv, acc[j]);
        }
        __syncthreads();          // done reading buf[cur] before it is re-filled
    }
#else
    for (int k0 = 0; k0 < K; k0 += 32) {
        {   // A tile: 64x32 bf16 = 256 x b128, one chunk per thread
            const int r = tid >> 2, c = (tid & 3) * 8;
            *(int4*)&Al[0][r][c] = *(const int4*)(A + (size_t)(m0 + r) * K + k0 + c);
        }
        #pragma unroll
        for (int t = 0; t < 2; ++t) {   // B tile: 128x32 = 512 x b128
            const int idx = tid * 2 + t;
            const int r = idx >> 2, c = (idx & 3) * 8;
            const int n = n0 + r;
            int4 val = {0, 0, 0, 0};
            if (n < N) val = *(const int4*)(W + (size_t)n * K + k0 + c);
            *(int4*)&Bl[0][r][c] = val;
        }
        __syncthreads();
        bf16x16 af = ldfrag(&Al[0][wm * 16 + l16][0], hlf * 8, 16 + hlf * 8);
        #pragma unroll
        for (int j = 0; j < 4; ++j) {
            bf16x16 bfv = ldfrag(&Bl[0][wn * 64 + j * 16 + l16][0],
                                 hlf * 8, 16 + hlf * 8);
            acc[j] = wmma_bf16(af, bfv, acc[j]);
        }
        __syncthreads();
    }
#endif

    #pragma unroll
    for (int j = 0; j < 4; ++j) {
        const int n = n0 + wn * 64 + j * 16 + l16;
        if (n < N) {
            const float bv = bias ? bias[n] : 0.0f;
            #pragma unroll
            for (int r = 0; r < 8; ++r) {
                const int m = m0 + wm * 16 + r + 8 * hlf;  // C row = r + 8*half
                C[(size_t)m * N + n] = (TOut)(acc[j][r] + bv);
            }
        }
    }
}

// ---------------------------------------------------------------------------
// Fused causal flash attention (bf16 in / bf16 out). Grid: (S/128, H, B).
// Each wave owns 16 query rows. Layout: [B*S, D], head h at cols [h*64,h*64+64).
// ---------------------------------------------------------------------------
__global__ void __launch_bounds__(256)
attn_kernel(const __bf16* __restrict__ Q, const __bf16* __restrict__ Kc,
            const __bf16* __restrict__ Vc, __bf16* __restrict__ Y) {
    __shared__ unsigned short Kl[32][64];    // 4 KB, keys row-major
    __shared__ unsigned short Vt[64][32];    // 4 KB, V transposed (dh major)
    __shared__ __bf16 Pl[8][16][32];         // 8 KB, per-wave P relayout

    const int qb   = blockIdx.x;
    const int h    = blockIdx.y;
    const int bb   = blockIdx.z;
    const int tid  = threadIdx.x;
    const int wav  = tid >> 5;
    const int lane = tid & 31;
    const int hlf  = lane >> 4;
    const int l16  = lane & 15;
    const int rowbase = qb * 128 + wav * 16;
    const int hoff = h * DH;

    // Q fragments (16x64 per wave) loaded straight from global bf16
    bf16x16 qf[2];
    {
        const __bf16* qrow = Q + ((size_t)(bb * S + rowbase + l16)) * D + hoff;
        #pragma unroll
        for (int s2 = 0; s2 < 2; ++s2)
            qf[s2] = ldfrag(qrow, s2 * 32 + hlf * 8, s2 * 32 + 16 + hlf * 8);
    }

    f32x8 O[4];
    #pragma unroll
    for (int t = 0; t < 4; ++t) O[t] = zero8();
    float mrow[8], lrow[8];
    #pragma unroll
    for (int r = 0; r < 8; ++r) { mrow[r] = -1e30f; lrow[r] = 0.0f; }

    const int jr = tid >> 3;        // staging: key row 0..31
    const int c8 = (tid & 7) * 8;   // dh col 0..56 (one b128 per thread)

    const int nkb = (qb + 1) * 4;   // causal: keys up to qb*128+127
    for (int kb = 0; kb < nkb; ++kb) {
        const int j0 = kb * 32;
        {
            const __bf16* krow = Kc + ((size_t)(bb * S + j0 + jr)) * D + hoff + c8;
            const __bf16* vrow = Vc + ((size_t)(bb * S + j0 + jr)) * D + hoff + c8;
            *(int4*)&Kl[jr][c8] = *(const int4*)krow;
            union { int4 q; unsigned short u[8]; } vv;
            vv.q = *(const int4*)vrow;
            #pragma unroll
            for (int i = 0; i < 8; ++i) Vt[c8 + i][jr] = vv.u[i];
        }
        __syncthreads();

        // S = Q K^T : 16x32 per wave as two 16x16 tiles, dh split in two K=32 steps
        f32x8 Sv[2];
        Sv[0] = zero8(); Sv[1] = zero8();
        #pragma unroll
        for (int t = 0; t < 2; ++t) {
            #pragma unroll
            for (int s2 = 0; s2 < 2; ++s2) {
                bf16x16 kf = ldfrag(&Kl[t * 16 + l16][0],
                                    s2 * 32 + hlf * 8, s2 * 32 + 16 + hlf * 8);
                Sv[t] = wmma_bf16(qf[s2], kf, Sv[t]);
            }
        }

        // online softmax; 1/sqrt(dh)=0.125 folded in here
        float alpha[8];
        #pragma unroll
        for (int r = 0; r < 8; ++r) {
            const int row = rowbase + r + 8 * hlf;
            float s0 = Sv[0][r] * 0.125f;
            float s1 = Sv[1][r] * 0.125f;
            if (j0 + l16 > row)      s0 = -1e30f;   // causal mask
            if (j0 + 16 + l16 > row) s1 = -1e30f;
            float mx = fmaxf(s0, s1);
            mx = fmaxf(mx, __shfl_xor(mx, 1, 32));
            mx = fmaxf(mx, __shfl_xor(mx, 2, 32));
            mx = fmaxf(mx, __shfl_xor(mx, 4, 32));
            mx = fmaxf(mx, __shfl_xor(mx, 8, 32));
            const float mnew = fmaxf(mrow[r], mx);
            const float al   = __expf(mrow[r] - mnew);
            const float p0   = __expf(s0 - mnew);
            const float p1   = __expf(s1 - mnew);
            float ps = p0 + p1;
            ps += __shfl_xor(ps, 1, 32);
            ps += __shfl_xor(ps, 2, 32);
            ps += __shfl_xor(ps, 4, 32);
            ps += __shfl_xor(ps, 8, 32);
            lrow[r]  = lrow[r] * al + ps;
            mrow[r]  = mnew;
            alpha[r] = al;
            Pl[wav][r + 8 * hlf][l16]      = (__bf16)p0;
            Pl[wav][r + 8 * hlf][16 + l16] = (__bf16)p1;
        }
        #pragma unroll
        for (int t = 0; t < 4; ++t) {
            #pragma unroll
            for (int r = 0; r < 8; ++r) O[t][r] *= alpha[r];
        }
        __syncthreads();   // P relayout visible; K reads done

        bf16x16 pf = ldfrag(&Pl[wav][l16][0], hlf * 8, 16 + hlf * 8);
        #pragma unroll
        for (int t = 0; t < 4; ++t) {
            bf16x16 vf = ldfrag(&Vt[t * 16 + l16][0], hlf * 8, 16 + hlf * 8);
            O[t] = wmma_bf16(pf, vf, O[t]);
        }
        __syncthreads();   // before next K/V staging
    }

    #pragma unroll
    for (int t = 0; t < 4; ++t) {
        const int col = hoff + t * 16 + l16;
        #pragma unroll
        for (int r = 0; r < 8; ++r) {
            const int row = rowbase + r + 8 * hlf;
            Y[((size_t)(bb * S + row)) * D + col] = (__bf16)(O[t][r] / lrow[r]);
        }
    }
}

// ---------------------------------------------------------------------------
// LayerNorm with unbiased std, eps added to std: (x-mu)/(sigma+eps)*g+b.
// TOut = float (residual path) or __bf16 (GEMM A-operand path).
// ---------------------------------------------------------------------------
template <typename TOut>
__global__ void __launch_bounds__(256)
layernorm_kernel(const float* __restrict__ in, const float* __restrict__ g,
                 const float* __restrict__ bvec, TOut* __restrict__ out, int ncols) {
    __shared__ float red[256];
    const int row = blockIdx.x;
    const int tid = threadIdx.x;
    const float* x = in + (size_t)row * ncols;

    float s = 0.0f;
    for (int c = tid; c < ncols; c += 256) s += x[c];
    red[tid] = s;
    __syncthreads();
    for (int off = 128; off > 0; off >>= 1) {
        if (tid < off) red[tid] += red[tid + off];
        __syncthreads();
    }
    const float mean = red[0] / (float)ncols;
    __syncthreads();

    float vv = 0.0f;
    for (int c = tid; c < ncols; c += 256) { float d = x[c] - mean; vv += d * d; }
    red[tid] = vv;
    __syncthreads();
    for (int off = 128; off > 0; off >>= 1) {
        if (tid < off) red[tid] += red[tid + off];
        __syncthreads();
    }
    const float var = red[0] / (float)(ncols - 1);
    const float inv = 1.0f / (sqrtf(var) + 1e-6f);
    TOut* o = out + (size_t)row * ncols;
    for (int c = tid; c < ncols; c += 256)
        o[c] = (TOut)((x[c] - mean) * inv * g[c] + bvec[c]);
}

// GEGLU with exact GELU: out = a * 0.5*g*(1+erf(g/sqrt(2)))
__global__ void __launch_bounds__(256)
geglu_kernel(const float* __restrict__ hb, float* __restrict__ out) {
    const int idx = blockIdx.x * 256 + threadIdx.x;    // over M*FF
    const int r = idx / FF;
    const int c = idx - r * FF;
    const float a = hb[(size_t)r * (2 * FF) + c];
    const float g = hb[(size_t)r * (2 * FF) + FF + c];
    const float gl = 0.5f * g * (1.0f + erff(g * 0.70710678118654752440f));
    out[idx] = a * gl;
}

__global__ void __launch_bounds__(256)
add_kernel(float* __restrict__ x, const float* __restrict__ yv) {
    const int idx = blockIdx.x * 256 + threadIdx.x;    // over M*D (divisible)
    x[idx] += yv[idx];
}

// x = emb[ids]*sqrt(D) + pos_enc
__global__ void __launch_bounds__(256)
embed_kernel(const int* __restrict__ ids, const float* __restrict__ emb,
             const float* __restrict__ pos, float* __restrict__ x) {
    const int idx = blockIdx.x * 256 + threadIdx.x;    // over M*D
    const int m = idx / D;
    const int c = idx - m * D;
    const int s = m & (S - 1);
    const int t = ids[m];
    x[idx] = emb[(size_t)t * D + c] * 22.627416997969522f + pos[(size_t)s * D + c];
}

// f32 -> bf16 cast, 2 elements/thread (packed convert)
__global__ void __launch_bounds__(256)
castf2b_kernel(const float* __restrict__ src, __bf16* __restrict__ dst) {
    const int i = blockIdx.x * 256 + threadIdx.x;
    float2 f = ((const float2*)src)[i];
    bf16x2 o;
    o.x = (__bf16)f.x;
    o.y = (__bf16)f.y;
    ((bf16x2*)dst)[i] = o;
}

} // anonymous namespace

extern "C" void kernel_launch(void* const* d_in, const int* in_sizes, int n_in,
                              void* d_out, int out_size, void* d_ws, size_t ws_size,
                              hipStream_t stream) {
    (void)in_sizes; (void)n_in; (void)out_size; (void)ws_size;
    const int*   ids   = (const int*)d_in[0];
    // d_in[1] = attention_mask (causal tril) -> recomputed analytically in attn_kernel
    const float* emb   = (const float*)d_in[2];
    const float* pos   = (const float*)d_in[3];
    const float* projw = (const float*)d_in[4 + 18 * LAYERS];
    const float* projb = (const float*)d_in[5 + 18 * LAYERS];

    constexpr size_t MD  = (size_t)M * D;
    constexpr size_t MFF = (size_t)M * FF;
    constexpr size_t LW  = 4 * (size_t)D * D + 2 * (size_t)FF * D + (size_t)D * FF;

    char* wp = (char*)d_ws;
    auto alloc = [&](size_t bytes) { char* p = wp; wp += bytes; return p; };
    float*  x     = (float*)alloc(MD * 4);
    float*  t32   = (float*)alloc(MD * 4);      // o-proj / fc2 outputs
    float*  t32b  = (float*)alloc(MD * 4);      // LN f32 output for residual
    float*  hbuf  = (float*)alloc(MFF * 2 * 4); // fc1 output
    float*  inter = (float*)alloc(MFF * 4);     // geglu output
    __bf16* nb    = (__bf16*)alloc(MD * 2);     // LN bf16 out (qkv / fc1 A)
    __bf16* qb    = (__bf16*)alloc(MD * 2);
    __bf16* kb2   = (__bf16*)alloc(MD * 2);
    __bf16* vb2   = (__bf16*)alloc(MD * 2);
    __bf16* yb    = (__bf16*)alloc(MD * 2);     // attention out
    __bf16* ib    = (__bf16*)alloc(MFF * 2);    // LN(inter) bf16 (fc2 A)
    __bf16* xb    = (__bf16*)alloc(MD * 2);     // cast of x for LM head
    __bf16* warena = (__bf16*)alloc(((size_t)LAYERS * LW + (size_t)V * D) * 2);

    const dim3 blk(256);
    auto cast = [&](const float* s, __bf16* dst, size_t n) {
        castf2b_kernel<<<dim3((unsigned)(n / 512)), blk, 0, stream>>>(s, dst);
    };

    // Pre-cast all weights to bf16 (once per launch; deterministic)
    for (int l = 0; l < LAYERS; ++l) {
        const float* const* p = (const float* const*)(d_in + 4 + 18 * l);
        __bf16* base = warena + (size_t)l * LW;
        cast(p[0],  base,                                   (size_t)D * D);      // wq
        cast(p[2],  base + 1 * (size_t)D * D,               (size_t)D * D);      // wk
        cast(p[4],  base + 2 * (size_t)D * D,               (size_t)D * D);      // wv
        cast(p[6],  base + 3 * (size_t)D * D,               (size_t)D * D);      // wo
        cast(p[10], base + 4 * (size_t)D * D,               2 * (size_t)FF * D); // fc1
        cast(p[11], base + 4 * (size_t)D * D + 2 * (size_t)FF * D,
             (size_t)D * FF);                                                    // fc2
    }
    __bf16* projwb = warena + (size_t)LAYERS * LW;
    cast(projw, projwb, (size_t)V * D);

    embed_kernel<<<dim3((M * D) / 256), blk, 0, stream>>>(ids, emb, pos, x);

    for (int l = 0; l < LAYERS; ++l) {
        const float* const* p = (const float* const*)(d_in + 4 + 18 * l);
        const float *bq = p[1], *bk = p[3], *bv = p[5], *bo = p[7];
        const float *alg = p[8], *alb = p[9];
        const float *flg = p[12], *flb = p[13];
        const float *r1g = p[14], *r1b = p[15];
        const float *r2g = p[16], *r2b = p[17];
        __bf16* base = warena + (size_t)l * LW;
        const __bf16 *wqb = base;
        const __bf16 *wkb = base + 1 * (size_t)D * D;
        const __bf16 *wvb = base + 2 * (size_t)D * D;
        const __bf16 *wob = base + 3 * (size_t)D * D;
        const __bf16 *f1b = base + 4 * (size_t)D * D;
        const __bf16 *f2b = base + 4 * (size_t)D * D + 2 * (size_t)FF * D;

        layernorm_kernel<__bf16><<<dim3(M), blk, 0, stream>>>(x, r1g, r1b, nb, D);
        gemm_kernel<__bf16><<<dim3(D / 128, M / 64), blk, 0, stream>>>(nb, wqb, bq, qb, D, D);
        gemm_kernel<__bf16><<<dim3(D / 128, M / 64), blk, 0, stream>>>(nb, wkb, bk, kb2, D, D);
        gemm_kernel<__bf16><<<dim3(D / 128, M / 64), blk, 0, stream>>>(nb, wvb, bv, vb2, D, D);
        attn_kernel<<<dim3(S / 128, H, B), blk, 0, stream>>>(qb, kb2, vb2, yb);
        gemm_kernel<float><<<dim3(D / 128, M / 64), blk, 0, stream>>>(yb, wob, bo, t32, D, D);
        layernorm_kernel<float><<<dim3(M), blk, 0, stream>>>(t32, alg, alb, t32b, D);
        add_kernel<<<dim3((M * D) / 256), blk, 0, stream>>>(x, t32b);

        layernorm_kernel<__bf16><<<dim3(M), blk, 0, stream>>>(x, r2g, r2b, nb, D);
        gemm_kernel<float><<<dim3((2 * FF) / 128, M / 64), blk, 0, stream>>>(
            nb, f1b, nullptr, hbuf, 2 * FF, D);
        geglu_kernel<<<dim3((M * FF) / 256), blk, 0, stream>>>(hbuf, inter);
        layernorm_kernel<__bf16><<<dim3(M), blk, 0, stream>>>(inter, flg, flb, ib, FF);
        gemm_kernel<float><<<dim3(D / 128, M / 64), blk, 0, stream>>>(
            ib, f2b, nullptr, t32, D, FF);
        add_kernel<<<dim3((M * D) / 256), blk, 0, stream>>>(x, t32);
    }

    // LM head: logits = x @ proj_w^T + proj_b  (N=50257, OOB-guarded)
    cast(x, xb, MD);
    gemm_kernel<float><<<dim3((V + 127) / 128, M / 64), blk, 0, stream>>>(
        xb, projwb, projb, (float*)d_out, V, D);
}
